// LSTM_VAE_15891378995817
// MI455X (gfx1250) — matrix operand order repaired and checked by reading, not compiled, not run on hardware
//
#include <hip/hip_runtime.h>
#include <cmath>

// MI455X (gfx1250, wave32) LSTM-VAE.
// Recurrent matmuls [16,64]x[64,256] via v_wmma_f32_16x16x32_f16.
// Wave w owns N-columns [w*16, w*16+16) of ALL FOUR gates, so i/f/g/o for
// each (m,j) land in the same lane's D-layout registers and the whole cell
// update is in-register (no gate LDS exchange, branch-free activations).
// Activations use raw v_exp_f32 / v_rcp_f32 (TRANS32, co-executes with WMMA).

typedef __attribute__((ext_vector_type(16))) _Float16 v16h;
typedef __attribute__((ext_vector_type(8)))  float    v8f;

#define B_ 4096
#define T_ 512
#define D_ 2
#define H_ 64
#define L_ 32

#define LOG2E 1.4426950408889634f

// sigmoid(x): saturates correctly (rcp(inf)=0 -> 0 ; rcp(1)=1).
__device__ __forceinline__ float fast_sigm(float x) {
    float e = __builtin_amdgcn_exp2f(-LOG2E * x);
    return __builtin_amdgcn_rcpf(1.0f + e);
}

// tanh(x) = 1 - 2/(exp(2x)+1). Branch-free, correct saturation at +/-inf.
__device__ __forceinline__ float fast_tanh(float x) {
    float e = __builtin_amdgcn_exp2f((2.0f * LOG2E) * x);
    return 1.0f - 2.0f * __builtin_amdgcn_rcpf(e + 1.0f);
}

__device__ __forceinline__ float fast_exp(float x) {
    return __builtin_amdgcn_exp2f(LOG2E * x);
}

// f16 A-matrix 16x32 fragment from LDS row-major [16][64] halves.
// Lanes 0-15: M=lane, K = kb..kb+15, kb=0;
// lanes 16-31: same M, kb=16.
__device__ __forceinline__ v16h load_a_frag(const _Float16* s, int lane, int kt) {
    int m  = lane & 15;
    int kb = kt * 32 + ((lane & 16) ? 16 : 0);
    const _Float16* p = s + m * H_ + kb;
    v16h a;
#pragma unroll
    for (int e = 0; e < 8; ++e) a[e] = p[e];
#pragma unroll
    for (int e = 0; e < 8; ++e) a[8 + e] = p[8 + e];
    return a;
}

// f16 B-matrix 32x16 fragment from global row-major W[n][64] (fp32 -> f16).
// Lanes 0-15: N=lane, K = kt*32 + 0..15 ; lanes 16-31: same N, K = kt*32+16..31.
__device__ __forceinline__ v16h load_b_frag(const float* W, int n, int kt, int lane) {
    int kb = kt * 32 + ((lane & 16) ? 16 : 0);
    const float* p = W + n * H_ + kb;
    v16h b;
#pragma unroll
    for (int e = 0; e < 16; ++e) b[e] = (_Float16)p[e];
    return b;
}

// ---------------------------------------------------------------------------
// Encoder: 512-step LSTM over x[B,T,2]; emits mu/logvar into d_out tail.
// ---------------------------------------------------------------------------
__global__ __launch_bounds__(128) void enc_kernel(
    const float* __restrict__ x,   const float* __restrict__ Wih,
    const float* __restrict__ Whh, const float* __restrict__ bias,
    const float* __restrict__ muW, const float* __restrict__ mub,
    const float* __restrict__ lvW, const float* __restrict__ lvb,
    float* __restrict__ out)
{
    __shared__ _Float16 s_h16[16 * H_];   // h staged for A-fragments
    __shared__ float    s_x[16 * 2];      // x_t tile
    __shared__ float    s_hf[16 * H_];    // final h (fp32)

    const int tid  = threadIdx.x;
    const int lane = tid & 31;
    const int w    = tid >> 5;            // wave owns cols jw..jw+15 of all gates
    const int jw   = w * 16;
    const int jn   = jw + (lane & 15);    // this lane's hidden column
    const int m0   = blockIdx.x * 16;     // batch tile base

    // Resident Whh B-fragments: gate g tile has n = g*64 + jn.
    v16h  BW[4][2];
    float wi0[4], wi1[4], be[4];
#pragma unroll
    for (int g = 0; g < 4; ++g) {
        int n = g * 64 + jn;
#pragma unroll
        for (int kt = 0; kt < 2; ++kt) BW[g][kt] = load_b_frag(Whh, n, kt, lane);
        wi0[g] = Wih[n * 2 + 0];
        wi1[g] = Wih[n * 2 + 1];
        be[g]  = bias[n];
    }

    const int moff = (lane & 16) ? 8 : 0;
    float creg[8], hlast[8];
#pragma unroll
    for (int r = 0; r < 8; ++r) { creg[r] = 0.f; hlast[r] = 0.f; }

    // init h staging to zero ([16][64] halves; 128 threads x 8 contiguous)
    {
        int em = tid >> 3, ej = (tid & 7) * 8;
#pragma unroll
        for (int q = 0; q < 8; ++q) s_h16[em * H_ + ej + q] = (_Float16)0.f;
    }
    if (tid < 32) {
        int m = tid >> 1, d = tid & 1;
        s_x[tid] = x[(size_t)(m0 + m) * (T_ * D_) + d];
    }
    __syncthreads();

#pragma unroll 1
    for (int t = 0; t < T_; ++t) {
        v16h aH0 = load_a_frag(s_h16, lane, 0);
        v16h aH1 = load_a_frag(s_h16, lane, 1);
        __syncthreads();   // loads done before anyone overwrites h staging

        float xv0[8], xv1[8];
#pragma unroll
        for (int r = 0; r < 8; ++r) {
            xv0[r] = s_x[(r + moff) * 2 + 0];
            xv1[r] = s_x[(r + moff) * 2 + 1];
        }

        // 4 gate tiles, same (m,j) block: 8 WMMAs total.
        v8f acc[4];
#pragma unroll
        for (int g = 0; g < 4; ++g) {
            v8f a;
#pragma unroll
            for (int r = 0; r < 8; ++r)
                a[r] = be[g] + xv0[r] * wi0[g] + xv1[r] * wi1[g];   // bias + x@Wih.T (K=2)
            a = __builtin_amdgcn_wmma_f32_16x16x32_f16(false, aH0, false, BW[g][0],
                                                       (short)0, a, false, false);
            a = __builtin_amdgcn_wmma_f32_16x16x32_f16(false, aH1, false, BW[g][1],
                                                       (short)0, a, false, false);
            acc[g] = a;
        }

        // In-register LSTM cell update (branch-free activations).
#pragma unroll
        for (int r = 0; r < 8; ++r) {
            float iv = fast_sigm(acc[0][r]);
            float fv = fast_sigm(acc[1][r]);
            float gv = fast_tanh(acc[2][r]);
            float ov = fast_sigm(acc[3][r]);
            float c  = fv * creg[r] + iv * gv;
            creg[r]  = c;
            float h  = ov * fast_tanh(c);
            hlast[r] = h;
            s_h16[(r + moff) * H_ + jn] = (_Float16)h;
        }
        __syncthreads();   // h staging complete for next step's A loads

        if (t + 1 < T_ && tid < 32) {
            int m = tid >> 1, d = tid & 1;
            s_x[tid] = x[(size_t)(m0 + m) * (T_ * D_) + (t + 1) * D_ + d];
        }
        if (t + 2 < T_ && tid < 32) {
            int m = tid >> 1, d = tid & 1;
            __builtin_prefetch(&x[(size_t)(m0 + m) * (T_ * D_) + (t + 2) * D_ + d], 0, 0);
        }
    }

    // mu / logvar projections from final h (fp32).
#pragma unroll
    for (int r = 0; r < 8; ++r) s_hf[(r + moff) * H_ + jn] = hlast[r];
    __syncthreads();

    const size_t MU_OFF = (size_t)B_ * T_ * D_;
    const size_t LV_OFF = MU_OFF + (size_t)B_ * L_;
#pragma unroll
    for (int rep = 0; rep < 4; ++rep) {
        int   idx = tid + rep * 128;
        int   m   = idx >> 5, l = idx & 31;
        float am = mub[l], al = lvb[l];
#pragma unroll 8
        for (int k = 0; k < H_; ++k) {
            float hv = s_hf[m * H_ + k];
            am += hv * muW[l * H_ + k];
            al += hv * lvW[l * H_ + k];
        }
        out[MU_OFF + (size_t)(m0 + m) * L_ + l] = am;
        out[LV_OFF + (size_t)(m0 + m) * L_ + l] = al;
    }
}

// ---------------------------------------------------------------------------
// Decoder: reparameterize (reads mu/logvar from d_out), then 512-step
// autoregressive LSTM with output feedback. 16 WMMAs/step/wave.
// ---------------------------------------------------------------------------
__global__ __launch_bounds__(128) void dec_kernel(
    const float* __restrict__ eps,
    const float* __restrict__ dinW, const float* __restrict__ dinb,
    const float* __restrict__ dWih, const float* __restrict__ dWhh,
    const float* __restrict__ db,
    const float* __restrict__ outW, const float* __restrict__ outb,
    const float* __restrict__ o2hW, const float* __restrict__ o2hb,
    float* __restrict__ out)
{
    __shared__ _Float16 s_h16[16 * H_];   // h staging (A layout source)
    __shared__ _Float16 s_i16[16 * H_];   // feedback input staging
    __shared__ float    s_o[16 * 2];      // out[m][d] this step
    __shared__ float    s_outW[2 * H_];
    __shared__ float    s_z[16 * L_];

    const int tid  = threadIdx.x;
    const int lane = tid & 31;
    const int w    = tid >> 5;
    const int jw   = w * 16;
    const int jn   = jw + (lane & 15);
    const int m0   = blockIdx.x * 16;
    const int moff = (lane & 16) ? 8 : 0;

    v16h  BH[4][2], BI[4][2];
    float be[4];
#pragma unroll
    for (int g = 0; g < 4; ++g) {
        int n = g * 64 + jn;
#pragma unroll
        for (int kt = 0; kt < 2; ++kt) {
            BH[g][kt] = load_b_frag(dWhh, n, kt, lane);
            BI[g][kt] = load_b_frag(dWih, n, kt, lane);
        }
        be[g] = db[n];
    }

    const int em = tid >> 3;
    const int ej = (tid & 7) * 8;
    float o2w0[8], o2w1[8], o2b[8];
#pragma unroll
    for (int q = 0; q < 8; ++q) {
        int j   = ej + q;
        o2w0[q] = o2hW[j * 2 + 0];
        o2w1[q] = o2hW[j * 2 + 1];
        o2b[q]  = o2hb[j];
    }
    s_outW[tid % 128] = outW[tid % 128];  // 2*64 floats staged once

    // z = mu + eps * exp(0.5*logvar)
    const size_t MU_OFF = (size_t)B_ * T_ * D_;
    const size_t LV_OFF = MU_OFF + (size_t)B_ * L_;
#pragma unroll
    for (int rep = 0; rep < 4; ++rep) {
        int   idx = tid + rep * 128;
        int   m   = idx >> 5, l = idx & 31;
        float muv = out[MU_OFF + (size_t)(m0 + m) * L_ + l];
        float lvv = out[LV_OFF + (size_t)(m0 + m) * L_ + l];
        float ev  = eps[(size_t)(m0 + m) * L_ + l];
        s_z[m * L_ + l] = muv + ev * fast_exp(0.5f * lvv);
    }
    __syncthreads();

    // hidden = z @ dec_in_W.T + b ; inp = 0
    {
#pragma unroll
        for (int q = 0; q < 8; ++q) {
            int   j   = ej + q;
            float acc = dinb[j];
#pragma unroll 8
            for (int l = 0; l < L_; ++l) acc += s_z[em * L_ + l] * dinW[j * L_ + l];
            s_h16[em * H_ + j] = (_Float16)acc;
            s_i16[em * H_ + j] = (_Float16)0.f;
        }
    }
    float creg[8];
#pragma unroll
    for (int r = 0; r < 8; ++r) creg[r] = 0.f;
    __syncthreads();

#pragma unroll 1
    for (int t = 0; t < T_; ++t) {
        v16h aH0 = load_a_frag(s_h16, lane, 0);
        v16h aH1 = load_a_frag(s_h16, lane, 1);
        v16h aI0 = load_a_frag(s_i16, lane, 0);
        v16h aI1 = load_a_frag(s_i16, lane, 1);
        __syncthreads();   // (1) frag loads done before h staging rewrite

        v8f acc[4];
#pragma unroll
        for (int g = 0; g < 4; ++g) {
            v8f a;
#pragma unroll
            for (int r = 0; r < 8; ++r) a[r] = be[g];
            a = __builtin_amdgcn_wmma_f32_16x16x32_f16(false, aI0, false, BI[g][0],
                                                       (short)0, a, false, false);
            a = __builtin_amdgcn_wmma_f32_16x16x32_f16(false, aI1, false, BI[g][1],
                                                       (short)0, a, false, false);
            a = __builtin_amdgcn_wmma_f32_16x16x32_f16(false, aH0, false, BH[g][0],
                                                       (short)0, a, false, false);
            a = __builtin_amdgcn_wmma_f32_16x16x32_f16(false, aH1, false, BH[g][1],
                                                       (short)0, a, false, false);
            acc[g] = a;
        }

        // In-register cell update.
#pragma unroll
        for (int r = 0; r < 8; ++r) {
            float iv = fast_sigm(acc[0][r]);
            float fv = fast_sigm(acc[1][r]);
            float gv = fast_tanh(acc[2][r]);
            float ov = fast_sigm(acc[3][r]);
            float c  = fv * creg[r] + iv * gv;
            creg[r]  = c;
            float h  = ov * fast_tanh(c);
            s_h16[(r + moff) * H_ + jn] = (_Float16)h;
        }
        __syncthreads();   // (2) h staging complete

        // out = h @ out_W.T + out_b  -> d_out and feedback
        if (tid < 32) {
            int   m = tid >> 1, d = tid & 1;
            float acc2 = outb[d];
#pragma unroll 8
            for (int k = 0; k < H_; ++k)
                acc2 += (float)s_h16[m * H_ + k] * s_outW[d * H_ + k];
            out[(size_t)(m0 + m) * (T_ * D_) + t * D_ + d] = acc2;
            s_o[m * 2 + d] = acc2;
        }
        __syncthreads();   // (3) s_o ready

        // nxt = out @ o2h_W.T + o2h_b  (K=2, scalar FMAs)
#pragma unroll
        for (int q = 0; q < 8; ++q) {
            int   j  = ej + q;
            float nv = o2b[q] + s_o[em * 2 + 0] * o2w0[q] + s_o[em * 2 + 1] * o2w1[q];
            s_i16[em * H_ + j] = (_Float16)nv;
        }
        __syncthreads();   // (4) i staging ready for next frag loads
    }
}

extern "C" void kernel_launch(void* const* d_in, const int* in_sizes, int n_in,
                              void* d_out, int out_size, void* d_ws, size_t ws_size,
                              hipStream_t stream) {
    (void)in_sizes; (void)n_in; (void)d_ws; (void)ws_size; (void)out_size;
    const float* x      = (const float*)d_in[0];
    const float* eps    = (const float*)d_in[1];
    const float* encWih = (const float*)d_in[2];
    const float* encWhh = (const float*)d_in[3];
    const float* encb   = (const float*)d_in[4];
    const float* muW    = (const float*)d_in[5];
    const float* mub    = (const float*)d_in[6];
    const float* lvW    = (const float*)d_in[7];
    const float* lvb    = (const float*)d_in[8];
    const float* dinW   = (const float*)d_in[9];
    const float* dinb   = (const float*)d_in[10];
    const float* dWih   = (const float*)d_in[11];
    const float* dWhh   = (const float*)d_in[12];
    const float* db     = (const float*)d_in[13];
    const float* outW   = (const float*)d_in[14];
    const float* outb   = (const float*)d_in[15];
    const float* o2hW   = (const float*)d_in[16];
    const float* o2hb   = (const float*)d_in[17];
    float* out = (float*)d_out;

    dim3 grid(B_ / 16), block(128);
    enc_kernel<<<grid, block, 0, stream>>>(x, encWih, encWhh, encb, muW, mub, lvW, lvb, out);
    dec_kernel<<<grid, block, 0, stream>>>(eps, dinW, dinb, dWih, dWhh, db,
                                           outW, outb, o2hW, o2hb, out);
}
